// MambaLayer_39591008535206
// MI455X (gfx1250) — compile-verified, matrix operand order checked by reading, not run on hardware
//
#include <hip/hip_runtime.h>
#include <hip/hip_bf16.h>

// ---- problem dims (hardcoded from setup_inputs) ----
#define BSZ   2
#define CDIM  256          // dim
#define LTOT  4096         // 16^3
#define DIN   512          // d_inner
#define DXZ   1024         // 2*d_inner
#define NST   16           // d_state
#define DTR   16           // dt_rank
#define KPROJ 48           // dt_rank + 2*d_state
#define MROWS (BSZ*LTOT)   // 8192
#define NCH   32           // scan chunks
#define CLEN  128          // LTOT / NCH

typedef __attribute__((ext_vector_type(8)))  _Float16 v8h;
typedef __attribute__((ext_vector_type(16))) _Float16 v16h;
typedef __attribute__((ext_vector_type(8)))  float    v8f;

__device__ __forceinline__ v16h ld16h(const _Float16* p) {
  v8h lo = *(const v8h*)(p);
  v8h hi = *(const v8h*)(p + 16);
  return __builtin_shufflevector(lo, hi, 0,1,2,3,4,5,6,7,8,9,10,11,12,13,14,15);
}

__device__ __forceinline__ float silu_f(float v)     { return v / (1.f + __expf(-v)); }
__device__ __forceinline__ float softplus_f(float v) { return v > 20.f ? v : log1pf(__expf(v)); }

// branch sequence-position -> memory-position map
__device__ __forceinline__ int seq2mem(int p, int br) {
  if (br == 0) return p;                       // forward
  if (br == 1) return (LTOT - 1) - p;          // backward (flip)
  return ((p & 15) << 8) | (p >> 4);           // slice: (p%16)*256 + p/16
}

// ---------------- fp32 -> fp16 convert ----------------
__global__ void cvt_f32f16_k(const float* __restrict__ in, _Float16* __restrict__ out, int n) {
  int i = blockIdx.x * blockDim.x + threadIdx.x;
  if (i < n) out[i] = (_Float16)in[i];
}

// ---------------- layernorm over C, output f16 (M=8192 x 256) ----------------
__global__ void layernorm_k(const float* __restrict__ x, const float* __restrict__ g,
                            const float* __restrict__ be, _Float16* __restrict__ xnh) {
  int idx = blockIdx.x * blockDim.x + threadIdx.x;   // b*L + l
  int b = idx >> 12;
  int l = idx & (LTOT - 1);
  const float* xp = x + (size_t)b * CDIM * LTOT + l;
  float s = 0.f, s2 = 0.f;
  for (int c = 0; c < CDIM; ++c) { float v = xp[(size_t)c * LTOT]; s += v; s2 += v * v; }
  float mu  = s  * (1.f / CDIM);
  float var = s2 * (1.f / CDIM) - mu * mu;
  float rs  = rsqrtf(var + 1e-5f);
  _Float16* o = xnh + (size_t)idx * CDIM;
  for (int c = 0; c < CDIM; ++c) {
    float v = xp[(size_t)c * LTOT];
    o[c] = (_Float16)(((v - mu) * rs) * g[c] + be[c]);
  }
}

// ---------------- WMMA GEMM: C[M,N] = A[M,K] * W[N,K]^T ----------------
// Template NT = number of 16-wide N tiles per wave (branch-free inner loop).
// Register double-buffering: next k-slab loads issue while current WMMAs run.
template <int NT>
__global__ void __launch_bounds__(256) wmma_gemm_nt(
    const _Float16* __restrict__ A, const _Float16* __restrict__ W,
    float* __restrict__ Cm, int M, int N, int K) {
  const int lane = threadIdx.x & 31;
  const int wave = threadIdx.x >> 5;
  const int sel  = lane >> 4;        // half-wave select
  const int l15  = lane & 15;
  const int m_base = blockIdx.y * 128 + wave * 16;
  const int n_base = blockIdx.x * (NT * 16);

  const _Float16* arow = A + (size_t)(m_base + l15) * K + sel * 8;
  const _Float16* wrow = W + (size_t)(n_base + l15) * K + sel * 8;
  const size_t wstride = (size_t)16 * K;

  v8f acc[NT];
  #pragma unroll
  for (int t = 0; t < NT; ++t) acc[t] = v8f{};

  v16h a_cur = ld16h(arow);
  v16h b_cur[NT];
  #pragma unroll
  for (int t = 0; t < NT; ++t) b_cur[t] = ld16h(wrow + (size_t)t * wstride);

  int kb = 0;
  #pragma unroll 2
  for (; kb < K - 32; kb += 32) {
    // issue next slab's loads first (in flight during WMMAs)
    v16h a_nxt = ld16h(arow + kb + 32);
    v16h b_nxt[NT];
    #pragma unroll
    for (int t = 0; t < NT; ++t) b_nxt[t] = ld16h(wrow + (size_t)t * wstride + kb + 32);
    __builtin_prefetch(arow + kb + 64, 0, 1);
    __builtin_prefetch(wrow + kb + 64, 0, 1);
    #pragma unroll
    for (int t = 0; t < NT; ++t)
      acc[t] = __builtin_amdgcn_wmma_f32_16x16x32_f16(false, a_cur, false, b_cur[t],
                                                      (short)0, acc[t], false, false);
    a_cur = a_nxt;
    #pragma unroll
    for (int t = 0; t < NT; ++t) b_cur[t] = b_nxt[t];
  }
  #pragma unroll
  for (int t = 0; t < NT; ++t)
    acc[t] = __builtin_amdgcn_wmma_f32_16x16x32_f16(false, a_cur, false, b_cur[t],
                                                    (short)0, acc[t], false, false);

  // C/D layout: lane(l15)=N, vgpr r -> M = r + sel*8
  float* crow = Cm + (size_t)(m_base + sel * 8) * N + n_base + l15;
  #pragma unroll
  for (int r = 0; r < 8; ++r) {
    #pragma unroll
    for (int t = 0; t < NT; ++t)
      crow[(size_t)r * N + t * 16] = acc[t][r];
  }
}

// ---------------- causal depthwise conv(4) + SiLU, branch-permuted input ----------------
__global__ void conv_silu_k(const float* __restrict__ xz, const float* __restrict__ cw,
                            const float* __restrict__ cb, _Float16* __restrict__ uh, int br) {
  int idx = blockIdx.x * blockDim.x + threadIdx.x;   // [0, MROWS*DIN)
  int d  = idx & (DIN - 1);
  int pl = idx >> 9;                 // b*L + p (sequence pos)
  int b  = pl >> 12;
  int p  = pl & (LTOT - 1);
  float acc = cb[d];
  #pragma unroll
  for (int j = 0; j < 4; ++j) {
    int q = p - 3 + j;
    if (q >= 0) {
      int lm = seq2mem(q, br);
      acc += cw[d * 4 + j] * xz[((size_t)(b * LTOT + lm) << 10) + d];
    }
  }
  uh[idx] = (_Float16)silu_f(acc);
}

// ---------------- delta = softplus(dt_raw @ dtw^T + dtb) ----------------
__global__ void dtproj_k(const float* __restrict__ xdbl, const float* __restrict__ dtw,
                         const float* __restrict__ dtb, float* __restrict__ delta) {
  int idx = blockIdx.x * blockDim.x + threadIdx.x;   // [0, MROWS*DIN)
  int d   = idx & (DIN - 1);
  int row = idx >> 9;
  float acc = dtb[d];
  const float* xr = xdbl + (size_t)row * KPROJ;      // first DTR entries = dt_raw
  const float* wr = dtw + d * DTR;
  #pragma unroll
  for (int r = 0; r < DTR; ++r) acc += xr[r] * wr[r];
  delta[idx] = softplus_f(acc);
}

// ---------------- chunked selective scan ----------------
// Pass 1: per chunk, decay product P[n] and chunk-local state h_loc[n] (h_in = 0).
__global__ void scan_pass1_k(const _Float16* __restrict__ uh, const float* __restrict__ delta,
                             const float* __restrict__ xdbl, const float* __restrict__ A_log,
                             float* __restrict__ Pbuf, float* __restrict__ Hloc) {
  int idx  = blockIdx.x * blockDim.x + threadIdx.x;  // [0, BSZ*NCH*DIN)
  int d    = idx & (DIN - 1);
  int rest = idx >> 9;
  int c    = rest & (NCH - 1);
  int b    = rest >> 5;
  float Av[NST], P[NST], h[NST];
  #pragma unroll
  for (int n = 0; n < NST; ++n) { Av[n] = -__expf(A_log[d * NST + n]); P[n] = 1.f; h[n] = 0.f; }
  const int p0 = c * CLEN;
  for (int t = 0; t < CLEN; ++t) {
    size_t row = (size_t)(b * LTOT + p0 + t);
    float dt = delta[(row << 9) + d];
    float uu = (float)uh[(row << 9) + d];
    const float* bc = xdbl + row * KPROJ + DTR;
    float du = dt * uu;
    #pragma unroll
    for (int n = 0; n < NST; ++n) {
      float dec = __expf(dt * Av[n]);
      P[n] *= dec;
      h[n] = h[n] * dec + du * bc[n];
    }
  }
  float* Pp = Pbuf + (size_t)idx * NST;
  float* Hp = Hloc + (size_t)idx * NST;
  #pragma unroll
  for (int n = 0; n < NST; ++n) { Pp[n] = P[n]; Hp[n] = h[n]; }
}

// Sequential combine across chunks: h_in(c) for every chunk.
__global__ void scan_combine_k(const float* __restrict__ Pbuf, const float* __restrict__ Hloc,
                               float* __restrict__ Hin) {
  int idx = blockIdx.x * blockDim.x + threadIdx.x;   // [0, BSZ*DIN)
  int d = idx & (DIN - 1);
  int b = idx >> 9;
  float h[NST];
  #pragma unroll
  for (int n = 0; n < NST; ++n) h[n] = 0.f;
  for (int c = 0; c < NCH; ++c) {
    size_t o = ((size_t)((b * NCH + c) * DIN + d)) * NST;
    #pragma unroll
    for (int n = 0; n < NST; ++n) {
      Hin[o + n] = h[n];
      h[n] = Pbuf[o + n] * h[n] + Hloc[o + n];
    }
  }
}

// Pass 2: replay each chunk from its true h_in; emit gated output.
__global__ void scan_pass2_k(const _Float16* __restrict__ uh, const float* __restrict__ delta,
                             const float* __restrict__ xdbl, const float* __restrict__ xz,
                             const float* __restrict__ A_log, const float* __restrict__ Dp,
                             const float* __restrict__ Hin, float* __restrict__ ytot, int br) {
  int idx  = blockIdx.x * blockDim.x + threadIdx.x;  // [0, BSZ*NCH*DIN)
  int d    = idx & (DIN - 1);
  int rest = idx >> 9;
  int c    = rest & (NCH - 1);
  int b    = rest >> 5;
  float Av[NST], h[NST];
  const float* hp = Hin + (size_t)idx * NST;
  #pragma unroll
  for (int n = 0; n < NST; ++n) { Av[n] = -__expf(A_log[d * NST + n]); h[n] = hp[n]; }
  float Dd = Dp[d];
  const int p0 = c * CLEN;
  for (int t = 0; t < CLEN; ++t) {
    int p = p0 + t;
    size_t row = (size_t)(b * LTOT + p);
    float dt = delta[(row << 9) + d];
    float uu = (float)uh[(row << 9) + d];
    const float* bc = xdbl + row * KPROJ + DTR;      // Bm[0..15], Cm[16..31]
    float du = dt * uu;
    float y  = Dd * uu;
    #pragma unroll
    for (int n = 0; n < NST; ++n) {
      h[n] = h[n] * __expf(dt * Av[n]) + du * bc[n];
      y += h[n] * bc[NST + n];
    }
    int lm  = seq2mem(p, br);
    float z = xz[((size_t)(b * LTOT + lm) << 10) + DIN + d];
    ytot[((size_t)(b * LTOT + lm) << 9) + d] += y * silu_f(z);
  }
}

// ---------------- residual + transpose back to (B, C, L) ----------------
__global__ void epilogue_k(const float* __restrict__ x, const float* __restrict__ proj,
                           float* __restrict__ out) {
  int idx = blockIdx.x * blockDim.x + threadIdx.x;   // b*C*L + c*L + l
  int l = idx & (LTOT - 1);
  int c = (idx >> 12) & (CDIM - 1);
  int b = idx >> 20;
  out[idx] = x[idx] + proj[((size_t)(b * LTOT + l) << 8) + c];
}

extern "C" void kernel_launch(void* const* d_in, const int* in_sizes, int n_in,
                              void* d_out, int out_size, void* d_ws, size_t ws_size,
                              hipStream_t stream) {
  (void)in_sizes; (void)n_in; (void)out_size; (void)ws_size;
  const float* x    = (const float*)d_in[0];
  const float* ln_g = (const float*)d_in[1];
  const float* ln_b = (const float*)d_in[2];
  const float* wi   = (const float*)d_in[3];   // (1024, 256)
  const float* wo   = (const float*)d_in[4];   // (256, 512)
  // d_in[5] = nslices (hardcoded 16)
  const float *cw[3], *cb[3], *xw[3], *dtw[3], *dtb[3], *Al[3], *Dp[3];
  for (int i = 0; i < 3; ++i) {
    int base = 6 + 7 * i;
    cw[i]  = (const float*)d_in[base + 0];
    cb[i]  = (const float*)d_in[base + 1];
    xw[i]  = (const float*)d_in[base + 2];
    dtw[i] = (const float*)d_in[base + 3];
    dtb[i] = (const float*)d_in[base + 4];
    Al[i]  = (const float*)d_in[base + 5];
    Dp[i]  = (const float*)d_in[base + 6];
  }

  // ---- deterministic workspace carve (256B aligned) ----
  char* wsb = (char*)d_ws;
  size_t off = 0;
  auto carve = [&](size_t bytes) -> void* {
    void* p = wsb + off;
    off += (bytes + 255) & ~(size_t)255;
    return p;
  };
  _Float16* xnh = (_Float16*)carve((size_t)MROWS * CDIM * 2);   // normalized input, f16
  _Float16* wih = (_Float16*)carve((size_t)DXZ * CDIM * 2);     // in_proj_w f16
  _Float16* woh = (_Float16*)carve((size_t)CDIM * DIN * 2);     // out_proj_w f16
  _Float16* xwh[3];
  for (int i = 0; i < 3; ++i) xwh[i] = (_Float16*)carve((size_t)KPROJ * DIN * 2);
  float*    xz    = (float*)carve((size_t)MROWS * DXZ * 4);     // (B*L, 1024)
  _Float16* uh    = (_Float16*)carve((size_t)MROWS * DIN * 2);  // conv+silu out, f16
  float*    xdbl  = (float*)carve((size_t)MROWS * KPROJ * 4);
  float*    delta = (float*)carve((size_t)MROWS * DIN * 4);
  float*    ytot  = (float*)carve((size_t)MROWS * DIN * 4);
  _Float16* yh    = (_Float16*)carve((size_t)MROWS * DIN * 2);
  float*    proj  = (float*)carve((size_t)MROWS * CDIM * 4);
  float*    Pbuf  = (float*)carve((size_t)BSZ * NCH * DIN * NST * 4);
  float*    Hloc  = (float*)carve((size_t)BSZ * NCH * DIN * NST * 4);
  float*    Hin   = (float*)carve((size_t)BSZ * NCH * DIN * NST * 4);

  // weights -> f16 (once per call, cheap)
  cvt_f32f16_k<<<(DXZ * CDIM + 255) / 256, 256, 0, stream>>>(wi, wih, DXZ * CDIM);
  cvt_f32f16_k<<<(CDIM * DIN + 255) / 256, 256, 0, stream>>>(wo, woh, CDIM * DIN);
  for (int i = 0; i < 3; ++i)
    cvt_f32f16_k<<<(KPROJ * DIN + 255) / 256, 256, 0, stream>>>(xw[i], xwh[i], KPROJ * DIN);

  // layernorm -> xnh
  layernorm_k<<<MROWS / 256, 256, 0, stream>>>(x, ln_g, ln_b, xnh);

  // in_proj: xz[8192,1024] = xnh[8192,256] * wih[1024,256]^T
  wmma_gemm_nt<4><<<dim3(DXZ / 64, MROWS / 128), 256, 0, stream>>>(xnh, wih, xz, MROWS, DXZ, CDIM);

  hipMemsetAsync(ytot, 0, (size_t)MROWS * DIN * 4, stream);

  for (int br = 0; br < 3; ++br) {
    conv_silu_k<<<(MROWS * DIN) / 256, 256, 0, stream>>>(xz, cw[br], cb[br], uh, br);
    // x_proj: xdbl[8192,48] = uh[8192,512] * xwh[48,512]^T  (3 tiles of 16)
    wmma_gemm_nt<3><<<dim3(1, MROWS / 128), 256, 0, stream>>>(uh, xwh[br], xdbl, MROWS, KPROJ, DIN);
    dtproj_k<<<(MROWS * DIN) / 256, 256, 0, stream>>>(xdbl, dtw[br], dtb[br], delta);
    // chunked scan: pass1 (parallel) -> combine (tiny sequential) -> pass2 (parallel)
    scan_pass1_k<<<(BSZ * NCH * DIN) / 256, 256, 0, stream>>>(uh, delta, xdbl, Al[br], Pbuf, Hloc);
    scan_combine_k<<<(BSZ * DIN) / 256, 256, 0, stream>>>(Pbuf, Hloc, Hin);
    scan_pass2_k<<<(BSZ * NCH * DIN) / 256, 256, 0, stream>>>(uh, delta, xdbl, xz, Al[br], Dp[br],
                                                             Hin, ytot, br);
  }

  cvt_f32f16_k<<<(MROWS * DIN + 255) / 256, 256, 0, stream>>>(ytot, yh, MROWS * DIN);
  // out_proj: proj[8192,256] = yh[8192,512] * woh[256,512]^T
  wmma_gemm_nt<4><<<dim3(CDIM / 64, MROWS / 128), 256, 0, stream>>>(yh, woh, proj, MROWS, CDIM, DIN);
  epilogue_k<<<(BSZ * CDIM * LTOT) / 256, 256, 0, stream>>>(x, proj, (float*)d_out);
}
